// InvariantCrossDecoder_69930657513593
// MI455X (gfx1250) — matrix operand order repaired and checked by reading, not compiled, not run on hardware
//
#include <hip/hip_runtime.h>
#include <hip/hip_bf16.h>
#include <math.h>

#define H    128
#define NA   4096
#define NQ   16384
#define KTOP 48

// LDS row strides in halves, chosen so lane-bank = l*stride/2 mod 64 is
// distinct for l=0..15 (conflict-free ds_load_b128 A-fragment reads):
//  - 168 halves = 84 dwords  (first edge layer, K padded 132->160, read cols<160)
//  - 424 halves = 212 dwords (out layer, K padded 385->416)
//  - 136 halves = 68 dwords  (all 128-wide intermediate layers)
#define STR0 168
#define STR1 136
#define STRO 424

typedef __attribute__((ext_vector_type(16))) _Float16 v16h;
typedef __attribute__((ext_vector_type(8)))  float    v8f;
typedef __attribute__((ext_vector_type(4)))  float    f32x4;

union Frag16 { v16h h; f32x4 f[2]; };

__device__ __forceinline__ float gelu_f(float x) {
  return 0.5f * x * (1.0f + erff(x * 0.70710678118654752f));
}

__device__ __forceinline__ unsigned long long umin64(unsigned long long a, unsigned long long b) {
  return a < b ? a : b;
}

// ---- WMMA fragment loaders -------------------------------------------------
// A (16x32 f16) per ISA: lanes 0-15 row M=lane, K 0..7 (v0-3) + 16..23 (v4-7);
// lanes 16-31 row M=lane-16, K 8..15 + 24..31.
__device__ __forceinline__ v16h load_a_frag(const _Float16* lds, int stride,
                                            int rowBase, int kBase, int lane) {
  int row = rowBase + (lane & 15);
  int k0  = kBase + ((lane >> 4) << 3);
  const _Float16* p = lds + row * stride + k0;
  Frag16 fr;
  fr.f[0] = *(const f32x4*)(p);
  fr.f[1] = *(const f32x4*)(p + 16);
  return fr.h;
}

// B (32x16 f16): lane N=lane&15, K (lane>>4)*8 + 0..7 then +16..23 (matches A).
// Weights pre-packed in this order: frag stride 512 halves, lane stride 16
// halves -> two global b128 loads per fragment (weights are L2 resident).
__device__ __forceinline__ v16h load_b_frag(const _Float16* pack, int frag, int lane) {
  const _Float16* p = pack + frag * 512 + lane * 16;
  Frag16 fr;
  fr.f[0] = *(const f32x4*)(p);
  fr.f[1] = *(const f32x4*)(p + 8);
  return fr.h;
}

// One full 16-row x 128-col layer for a wave: K-loop of v_wmma_f32_16x16x32_f16,
// bias + optional exact-erf GELU fused into the D->LDS(f16) writeback.
// C/D layout: lane (M>=8?16:0)+N, vgpr r = M&7.
__device__ __forceinline__ void gemm_layer(const _Float16* Xin, int strideIn,
                                           _Float16* Xout, int strideOut,
                                           const _Float16* packW, const float* bias,
                                           int nKT, int rowBase, int lane, bool act) {
  const int n  = lane & 15;
  const int mb = (lane >> 4) << 3;
  for (int nt = 0; nt < 8; ++nt) {
    v8f acc = {};
    for (int kt = 0; kt < nKT; ++kt) {
      v16h a = load_a_frag(Xin, strideIn, rowBase, kt * 32, lane);
      v16h b = load_b_frag(packW, kt * 8 + nt, lane);
      acc = __builtin_amdgcn_wmma_f32_16x16x32_f16(false, a, false, b, (short)0, acc,
                                                   false, false);
    }
    const int col = nt * 16 + n;
    const float bv = bias[col];
#pragma unroll
    for (int r = 0; r < 8; ++r) {
      float v = acc[r] + bv;
      if (act) v = gelu_f(v);
      Xout[(rowBase + mb + r) * strideOut + col] = (_Float16)v;
    }
  }
}

// ---- weight pre-pack: fp32 (din x dout) -> f16 B-fragment stream -----------
__global__ void pack_w_kernel(const float* __restrict__ W, _Float16* __restrict__ P,
                              int din, int dout, int kpad, int npad) {
  const int total  = kpad * npad;
  const int ntiles = npad >> 4;
  for (int e = blockIdx.x * blockDim.x + threadIdx.x; e < total;
       e += gridDim.x * blockDim.x) {
    int frag   = e >> 9;
    int within = e & 511;
    int lane   = within >> 4;
    int j      = within & 15;
    int kt     = frag / ntiles;
    int nt     = frag - kt * ntiles;
    int K      = kt * 32 + ((lane >> 4) << 3) + (j & 7) + ((j >> 3) << 4);
    int N      = nt * 16 + (lane & 15);
    float v = (K < din && N < dout) ? W[K * dout + N] : 0.0f;
    P[e] = (_Float16)v;
  }
}

// ---- flow-direction dot products (u normalized on the fly) -----------------
__global__ void dotu_kernel(const float* __restrict__ X, const float* __restrict__ u3,
                            float* __restrict__ out, int n) {
  int i = blockIdx.x * blockDim.x + threadIdx.x;
  if (i >= n) return;
  float ux = u3[0], uy = u3[1], uz = u3[2];
  float inv = 1.0f / (sqrtf(ux * ux + uy * uy + uz * uz) + 1e-8f);
  out[i] = (X[i * 3] * ux + X[i * 3 + 1] * uy + X[i * 3 + 2] * uz) * inv;
}

// ---- global context: mean over anchors -> 3-layer MLP (one block) ----------
__global__ void gctx_kernel(const float* __restrict__ h,
                            const float* w0, const float* b0,
                            const float* w1, const float* b1,
                            const float* w2, const float* b2,
                            float* __restrict__ gctx) {
  __shared__ float x[H], y[H];
  int t = threadIdx.x;  // 128 threads
  float s = 0.0f;
  for (int i = 0; i < NA; ++i) s += h[(long)i * H + t];
  x[t] = s * (1.0f / (float)NA);
  __syncthreads();
  float acc = b0[t];
  for (int k = 0; k < H; ++k) acc += x[k] * w0[k * H + t];
  y[t] = gelu_f(acc);
  __syncthreads();
  acc = b1[t];
  for (int k = 0; k < H; ++k) acc += y[k] * w1[k * H + t];
  x[t] = gelu_f(acc);
  __syncthreads();
  acc = b2[t];
  for (int k = 0; k < H; ++k) acc += x[k] * w2[k * H + t];
  gctx[t] = acc;
}

// ---- top-48 nearest anchors per query (one block per query) ----------------
// d^2 row lives in LDS; 48 rounds of packed (f32bits<<32|idx) min-reduction
// produce ascending order with low-index tie-break (matches top_k semantics).
__global__ void topk_kernel(const float* __restrict__ A, const float* __restrict__ Q,
                            int* __restrict__ tk_idx, float* __restrict__ tk_d2) {
  __shared__ float d2[NA];
  __shared__ unsigned long long red[128];
  const int q = blockIdx.x, t = threadIdx.x;
  const float qx = Q[q * 3], qy = Q[q * 3 + 1], qz = Q[q * 3 + 2];
  for (int i = t; i < NA; i += 128) {
    float dx = qx - A[i * 3], dy = qy - A[i * 3 + 1], dz = qz - A[i * 3 + 2];
    d2[i] = dx * dx + dy * dy + dz * dz;
  }
  __syncthreads();
  for (int r = 0; r < KTOP; ++r) {
    unsigned long long best = ~0ull;
    for (int i = t; i < NA; i += 128) {
      unsigned long long c =
          (((unsigned long long)__float_as_uint(d2[i])) << 32) | (unsigned)i;
      best = umin64(best, c);
    }
    red[t] = best;
    __syncthreads();
    for (int s = 64; s > 0; s >>= 1) {
      if (t < s) red[t] = umin64(red[t], red[t + s]);
      __syncthreads();
    }
    if (t == 0) {
      unsigned long long w = red[0];
      int wi = (int)(w & 0xffffffffu);
      tk_idx[(long)q * KTOP + r] = wi;
      tk_d2[(long)q * KTOP + r]  = __uint_as_float((unsigned)(w >> 32));
      d2[wi] = __uint_as_float(0x7f800000u);  // +inf: remove from next rounds
    }
    __syncthreads();
  }
}

// ---- edge branch: gather -> 3-layer edge MLP -> gate -> softmax-agg --------
// EPB edges per block (multiple of KSLOT and of 16); each wave owns 16 rows and
// runs all GEMM layers barrier-free (rows disjoint across waves, buffers
// ping-pong A<->B).
template <int KSLOT, int EPB>
__global__ void branch_kernel(const float* __restrict__ hfeat,
                              const int* __restrict__ tk_idx,
                              const float* __restrict__ tk_d2,
                              const float* __restrict__ qdu,
                              const float* __restrict__ adu,
                              const _Float16* pw0, const float* pb0,
                              const _Float16* pw1, const float* pb1,
                              const _Float16* pw2, const float* pb2,
                              const _Float16* pg0, const float* gb0,
                              const _Float16* pg1, const float* gb1,
                              float* __restrict__ agg) {
  __shared__ _Float16 bufA[EPB * STR0];
  __shared__ _Float16 bufB[EPB * STR1];
  __shared__ float sLogit[EPB];
  __shared__ float sD2[EPB];
  __shared__ float sQv[EPB];
  __shared__ float sAv[EPB];
  __shared__ int   sNbr[EPB];

  const int t = threadIdx.x, nthr = blockDim.x;
  const long eBase = (long)blockIdx.x * EPB;

  for (int r = t; r < EPB; r += nthr) {
    long e   = eBase + r;
    int  q   = (int)(e / KSLOT);
    int slot = (int)(e - (long)q * KSLOT);
    int nb   = tk_idx[(long)q * KTOP + slot];
    sNbr[r] = nb;
    sD2[r]  = tk_d2[(long)q * KTOP + slot];
    sQv[r]  = qdu[q];
    sAv[r]  = adu[nb];
  }
  __syncthreads();

  // X0 (f16, K padded 132->160, stride 168): [neigh(128), d2, q.u, a.u, rel.u, 0..]
  for (int i = t; i < EPB * STR0; i += nthr) {
    int r = i / STR0, c = i - r * STR0;
    float v;
    if (c < H)           v = hfeat[(long)sNbr[r] * H + c];
    else if (c == H)     v = sD2[r];
    else if (c == H + 1) v = sQv[r];
    else if (c == H + 2) v = sAv[r];
    else if (c == H + 3) v = sQv[r] - sAv[r];
    else                 v = 0.0f;
    bufA[i] = (_Float16)v;
  }
  __syncthreads();

  const int lane    = t & 31;
  const int rowBase = (t >> 5) * 16;

  gemm_layer(bufA, STR0, bufB, STR1, pw0, pb0, 5, rowBase, lane, true);   // X1
  gemm_layer(bufB, STR1, bufA, STR1, pw1, pb1, 4, rowBase, lane, true);   // X2
  gemm_layer(bufA, STR1, bufB, STR1, pw2, pb2, 4, rowBase, lane, false);  // EH
  gemm_layer(bufB, STR1, bufA, STR1, pg0, gb0, 4, rowBase, lane, true);   // gate hidden

  {  // gate final: 128 -> 1 (N padded to 16, only column 0 is live)
    v8f acc = {};
    for (int kt = 0; kt < 4; ++kt) {
      v16h a = load_a_frag(bufA, STR1, rowBase, kt * 32, lane);
      v16h b = load_b_frag(pg1, kt, lane);
      acc = __builtin_amdgcn_wmma_f32_16x16x32_f16(false, a, false, b, (short)0, acc,
                                                   false, false);
    }
    if ((lane & 15) == 0) {
      int mb = (lane >> 4) << 3;
#pragma unroll
      for (int r = 0; r < 8; ++r) {
        int row = rowBase + mb + r;
        sLogit[row] = acc[r] + gb1[0] - sD2[row];  // distance-biased logit
      }
    }
  }
  __syncthreads();

  // softmax over k + weighted sum of EH rows -> agg[q]
  constexpr int QPB = EPB / KSLOT;
  for (int c = t; c < QPB * H; c += nthr) {
    int qq = c / H, col = c - qq * H;
    int base = qq * KSLOT;
    float mx = -3.0e38f;
    for (int s = 0; s < KSLOT; ++s) mx = fmaxf(mx, sLogit[base + s]);
    float den = 0.0f, num = 0.0f;
    for (int s = 0; s < KSLOT; ++s) {
      float w = __expf(sLogit[base + s] - mx);
      den += w;
      num += w * (float)bufB[(base + s) * STR1 + col];
    }
    long q = eBase / KSLOT + qq;
    agg[q * H + col] = num / den;
  }
}

// ---- output MLP: concat(agg_s, agg_l, g_ctx, qdu) padded 385->416 ----------
template <int RPB>
__global__ void out_kernel(const float* __restrict__ agg_s,
                           const float* __restrict__ agg_l,
                           const float* __restrict__ gctx,
                           const float* __restrict__ qdu,
                           const _Float16* pw0, const float* pb0,
                           const _Float16* pw1, const float* pb1,
                           const _Float16* pw2, const float* pb2,
                           float* __restrict__ out) {
  __shared__ _Float16 bufA[RPB * STRO];
  __shared__ _Float16 bufB[RPB * STR1];
  const int t = threadIdx.x, nthr = blockDim.x;
  const long rBase = (long)blockIdx.x * RPB;

  for (int i = t; i < RPB * STRO; i += nthr) {
    int r = i / STRO, c = i - r * STRO;
    long q = rBase + r;
    float v;
    if (c < 128)       v = agg_s[q * 128 + c];
    else if (c < 256)  v = agg_l[q * 128 + (c - 128)];
    else if (c < 384)  v = gctx[c - 256];
    else if (c == 384) v = qdu[q];
    else               v = 0.0f;
    bufA[i] = (_Float16)v;
  }
  __syncthreads();

  const int lane    = t & 31;
  const int rowBase = (t >> 5) * 16;
  gemm_layer(bufA, STRO, bufB, STR1, pw0, pb0, 13, rowBase, lane, true);
  gemm_layer(bufB, STR1, bufA, STR1, pw1, pb1, 4, rowBase, lane, true);

  v8f acc = {};
  for (int kt = 0; kt < 4; ++kt) {
    v16h a = load_a_frag(bufA, STR1, rowBase, kt * 32, lane);
    v16h b = load_b_frag(pw2, kt, lane);
    acc = __builtin_amdgcn_wmma_f32_16x16x32_f16(false, a, false, b, (short)0, acc,
                                                 false, false);
  }
  const int n = lane & 15, mb = (lane >> 4) << 3;
  if (n < 3) {
#pragma unroll
    for (int r = 0; r < 8; ++r) {
      long row = rBase + rowBase + mb + r;
      out[row * 3 + n] = acc[r] + pb2[n];
    }
  }
}

// ---------------------------------------------------------------------------
extern "C" void kernel_launch(void* const* d_in, const int* in_sizes, int n_in,
                              void* d_out, int out_size, void* d_ws, size_t ws_size,
                              hipStream_t stream) {
  (void)n_in; (void)out_size; (void)ws_size;
  const float* a_coords = (const float*)d_in[0];
  const float* h_feat   = (const float*)d_in[1];
  const float* q_coords = (const float*)d_in[2];
  const float* u_dir    = (const float*)d_in[3];

  // Params may be flattened in dict-insertion order (edge_s, gate_s, edge_l,
  // gate_l, global, out) or sorted-key order (edge_l, edge_s, gate_l, gate_s,
  // global, out). Disambiguate via in_sizes[10]: 132*128 => sorted.
  int iES, iGS, iEL, iGL;
  if (in_sizes[10] == 132 * 128) { iEL = 4; iES = 10; iGL = 16; iGS = 20; }
  else                           { iES = 4; iGS = 10; iEL = 14; iGL = 20; }
  const int iGlob = 24, iOut = 30;

  const float* es_w0 = (const float*)d_in[iES + 0]; const float* es_b0 = (const float*)d_in[iES + 1];
  const float* es_w1 = (const float*)d_in[iES + 2]; const float* es_b1 = (const float*)d_in[iES + 3];
  const float* es_w2 = (const float*)d_in[iES + 4]; const float* es_b2 = (const float*)d_in[iES + 5];
  const float* gs_w0 = (const float*)d_in[iGS + 0]; const float* gs_b0 = (const float*)d_in[iGS + 1];
  const float* gs_w1 = (const float*)d_in[iGS + 2]; const float* gs_b1 = (const float*)d_in[iGS + 3];
  const float* el_w0 = (const float*)d_in[iEL + 0]; const float* el_b0 = (const float*)d_in[iEL + 1];
  const float* el_w1 = (const float*)d_in[iEL + 2]; const float* el_b1 = (const float*)d_in[iEL + 3];
  const float* el_w2 = (const float*)d_in[iEL + 4]; const float* el_b2 = (const float*)d_in[iEL + 5];
  const float* gl_w0 = (const float*)d_in[iGL + 0]; const float* gl_b0 = (const float*)d_in[iGL + 1];
  const float* gl_w1 = (const float*)d_in[iGL + 2]; const float* gl_b1 = (const float*)d_in[iGL + 3];
  const float* g_w0  = (const float*)d_in[iGlob + 0]; const float* g_b0 = (const float*)d_in[iGlob + 1];
  const float* g_w1  = (const float*)d_in[iGlob + 2]; const float* g_b1 = (const float*)d_in[iGlob + 3];
  const float* g_w2  = (const float*)d_in[iGlob + 4]; const float* g_b2 = (const float*)d_in[iGlob + 5];
  const float* o_w0  = (const float*)d_in[iOut + 0];  const float* o_b0 = (const float*)d_in[iOut + 1];
  const float* o_w1  = (const float*)d_in[iOut + 2];  const float* o_b1 = (const float*)d_in[iOut + 3];
  const float* o_w2  = (const float*)d_in[iOut + 4];  const float* o_b2 = (const float*)d_in[iOut + 5];

  char* ws = (char*)d_ws;
  size_t off = 0;
  auto alloc = [&](size_t bytes) -> char* {
    char* p = ws + off;
    off = (off + bytes + 255) & ~(size_t)255;
    return p;
  };
  int*      tk_idx = (int*)alloc((size_t)NQ * KTOP * 4);
  float*    tk_d2  = (float*)alloc((size_t)NQ * KTOP * 4);
  float*    qdu    = (float*)alloc((size_t)NQ * 4);
  float*    adu    = (float*)alloc((size_t)NA * 4);
  float*    gctx   = (float*)alloc(128 * 4);
  float*    agg_s  = (float*)alloc((size_t)NQ * 128 * 4);
  float*    agg_l  = (float*)alloc((size_t)NQ * 128 * 4);
  _Float16* pes0 = (_Float16*)alloc(160 * 128 * 2);
  _Float16* pes1 = (_Float16*)alloc(128 * 128 * 2);
  _Float16* pes2 = (_Float16*)alloc(128 * 128 * 2);
  _Float16* pgs0 = (_Float16*)alloc(128 * 128 * 2);
  _Float16* pgs1 = (_Float16*)alloc(128 * 16 * 2);
  _Float16* pel0 = (_Float16*)alloc(160 * 128 * 2);
  _Float16* pel1 = (_Float16*)alloc(128 * 128 * 2);
  _Float16* pel2 = (_Float16*)alloc(128 * 128 * 2);
  _Float16* pgl0 = (_Float16*)alloc(128 * 128 * 2);
  _Float16* pgl1 = (_Float16*)alloc(128 * 16 * 2);
  _Float16* po0  = (_Float16*)alloc(416 * 128 * 2);
  _Float16* po1  = (_Float16*)alloc(128 * 128 * 2);
  _Float16* po2  = (_Float16*)alloc(128 * 16 * 2);

  auto packW = [&](const float* W, _Float16* P, int din, int dout, int kpad, int npad) {
    int total = kpad * npad;
    pack_w_kernel<<<(total + 255) / 256, 256, 0, stream>>>(W, P, din, dout, kpad, npad);
  };
  packW(es_w0, pes0, 132, 128, 160, 128);
  packW(es_w1, pes1, 128, 128, 128, 128);
  packW(es_w2, pes2, 128, 128, 128, 128);
  packW(gs_w0, pgs0, 128, 128, 128, 128);
  packW(gs_w1, pgs1, 128, 1,   128, 16);
  packW(el_w0, pel0, 132, 128, 160, 128);
  packW(el_w1, pel1, 128, 128, 128, 128);
  packW(el_w2, pel2, 128, 128, 128, 128);
  packW(gl_w0, pgl0, 128, 128, 128, 128);
  packW(gl_w1, pgl1, 128, 1,   128, 16);
  packW(o_w0,  po0,  385, 128, 416, 128);
  packW(o_w1,  po1,  128, 128, 128, 128);
  packW(o_w2,  po2,  128, 3,   128, 16);

  dotu_kernel<<<(NQ + 255) / 256, 256, 0, stream>>>(q_coords, u_dir, qdu, NQ);
  dotu_kernel<<<(NA + 255) / 256, 256, 0, stream>>>(a_coords, u_dir, adu, NA);
  gctx_kernel<<<1, 128, 0, stream>>>(h_feat, g_w0, g_b0, g_w1, g_b1, g_w2, g_b2, gctx);
  topk_kernel<<<NQ, 128, 0, stream>>>(a_coords, q_coords, tk_idx, tk_d2);

  // small branch: k=16, 64 edges (4 queries) per block, 4 waves
  branch_kernel<16, 64><<<(NQ * 16) / 64, 128, 0, stream>>>(
      h_feat, tk_idx, tk_d2, qdu, adu,
      pes0, es_b0, pes1, es_b1, pes2, es_b2, pgs0, gs_b0, pgs1, gs_b1, agg_s);
  // large branch: k=48, 48 edges (1 query) per block, 3 waves
  branch_kernel<48, 48><<<NQ, 96, 0, stream>>>(
      h_feat, tk_idx, tk_d2, qdu, adu,
      pel0, el_b0, pel1, el_b1, pel2, el_b2, pgl0, gl_b0, pgl1, gl_b1, agg_l);

  out_kernel<32><<<NQ / 32, 64, 0, stream>>>(
      agg_s, agg_l, gctx, qdu, po0, o_b0, po1, o_b1, po2, o_b2, (float*)d_out);
}
